// Self_Attention_26645977105087
// MI455X (gfx1250) — compile-verified, matrix operand order checked by reading, not run on hardware
//
#include <hip/hip_runtime.h>

typedef __attribute__((ext_vector_type(16))) __bf16 v16bf;
typedef __attribute__((ext_vector_type(8)))  __bf16 v8bf;
typedef __attribute__((ext_vector_type(8)))  float  v8f;
typedef __attribute__((ext_vector_type(4)))  float  f4;

#define DIM 1024
#define DK  128
#define SEQ 4096
#define NB  4
#define ROWS (NB*SEQ)      // 16384
#define FA_WAVES 4

// ---------- fragment helpers ----------

// A/B 16-bit fragment: elements [0..7] from p, [8..15] from p+16 (per ISA 7.12.2
// 16-bit 16x32 layout: lane-half kh selects K offsets {0..7,16..23} vs {8..15,24..31})
__device__ __forceinline__ v16bf load_bf16_frag(const __bf16* p) {
  v8bf lo = *(const v8bf*)p;          // 16B -> global_load_b128 / ds_load_b128
  v8bf hi = *(const v8bf*)(p + 16);
  v16bf r;
#pragma unroll
  for (int i = 0; i < 8; ++i) { r[i] = lo[i]; r[i + 8] = hi[i]; }
  return r;
}

// Same fragment but sourced from fp32 memory, converted to bf16 in registers.
__device__ __forceinline__ v16bf cvt_f32_frag(const float* p) {
  f4 a0 = *(const f4*)(p);      f4 a1 = *(const f4*)(p + 4);
  f4 b0 = *(const f4*)(p + 16); f4 b1 = *(const f4*)(p + 20);
  v16bf r;
#pragma unroll
  for (int i = 0; i < 4; ++i) {
    r[i]      = (__bf16)a0[i];
    r[i + 4]  = (__bf16)a1[i];
    r[i + 8]  = (__bf16)b0[i];
    r[i + 12] = (__bf16)b1[i];
  }
  return r;
}

__device__ __forceinline__ v8f zero8() {
  v8f z;
#pragma unroll
  for (int i = 0; i < 8; ++i) z[i] = 0.0f;
  return z;
}

// ---------- kernel 0: weight fp32 [DIM][DK] -> bf16 transposed [DK][DIM] ----------
__global__ void __launch_bounds__(256)
wcvt_kernel(const float* __restrict__ W, __bf16* __restrict__ Wt) {
  int idx = blockIdx.x * blockDim.x + threadIdx.x;   // DK*DIM threads
  int n = idx >> 10;          // output row (DK dim)
  int k = idx & (DIM - 1);    // output col (DIM dim)
  Wt[idx] = (__bf16)W[k * DK + n];
}

// ---------- kernel 1: QKV projection via bf16 WMMA ----------
// grid = (ROWS/128, 3), block = 256 (8 waves, each owns a 16x128 tile)
__global__ void __launch_bounds__(256)
qkv_proj_kernel(const float* __restrict__ X,
                const __bf16* __restrict__ WtAll,
                const float* __restrict__ bq, const float* __restrict__ bk,
                const float* __restrict__ bv,
                __bf16* __restrict__ Qb, __bf16* __restrict__ Kb,
                __bf16* __restrict__ Vt) {
  const int mat  = blockIdx.y;
  const int wave = threadIdx.x >> 5;
  const int lane = threadIdx.x & 31;
  const int n    = lane & 15;
  const int kh   = lane >> 4;
  const int r0   = blockIdx.x * 128 + wave * 16;

  const __bf16* Wt  = WtAll + (size_t)mat * (DK * DIM);
  const float* bias = (mat == 0) ? bq : (mat == 1) ? bk : bv;

  v8f acc[8];
#pragma unroll
  for (int t = 0; t < 8; ++t) acc[t] = zero8();

  const float* xrow = X + (size_t)(r0 + n) * DIM;

  for (int kk = 0; kk < DIM; kk += 32) {
    v16bf a = cvt_f32_frag(xrow + kk + kh * 8);
#pragma unroll
    for (int t = 0; t < 8; ++t) {
      v16bf bfrag = load_bf16_frag(Wt + (size_t)(t * 16 + n) * DIM + kk + kh * 8);
      acc[t] = __builtin_amdgcn_wmma_f32_16x16x32_bf16(
          false, a, false, bfrag, (short)0, acc[t], false, false);
    }
  }

  const int b  = r0 / SEQ;
  const int s0 = (r0 % SEQ) + (kh ? 8 : 0);

  if (mat < 2) {
    // Q / K: row-major bf16 [ROWS][DK]
    __bf16* O = (mat == 0) ? Qb : Kb;
#pragma unroll
    for (int t = 0; t < 8; ++t) {
      float bb = bias[t * 16 + n];
#pragma unroll
      for (int g = 0; g < 8; ++g) {
        int row = r0 + g + (kh ? 8 : 0);
        O[(size_t)row * DK + t * 16 + n] = (__bf16)(acc[t][g] + bb);
      }
    }
  } else {
    // V: transposed bf16 [NB][DK][SEQ]; 8 consecutive M values per lane -> one b128 store
#pragma unroll
    for (int t = 0; t < 8; ++t) {
      float bb = bias[t * 16 + n];
      v8bf pk;
#pragma unroll
      for (int g = 0; g < 8; ++g) pk[g] = (__bf16)(acc[t][g] + bb);
      *(v8bf*)(Vt + ((size_t)(b * DK + t * 16 + n)) * SEQ + s0) = pk;
    }
  }
}

// ---------- kernel 2: flash attention ----------
// grid = (SEQ/(FA_WAVES*16), NB), block = FA_WAVES*32. Each wave: 16 query rows,
// loops over keys in chunks of 32, online softmax in exp2 domain.
__global__ void __launch_bounds__(FA_WAVES * 32)
flash_attn_kernel(const __bf16* __restrict__ Qb, const __bf16* __restrict__ Kb,
                  const __bf16* __restrict__ Vt, float* __restrict__ Out) {
  __shared__ __bf16 lds_p[FA_WAVES][16][32];   // wave-private P transpose buffer

  const int wave = threadIdx.x >> 5;
  const int lane = threadIdx.x & 31;
  const int n    = lane & 15;
  const int kh   = lane >> 4;
  const int b    = blockIdx.y;
  const int q0   = blockIdx.x * (FA_WAVES * 16) + wave * 16;

  const float sc = 0.08838834764831845f * 1.4426950408889634f;  // dk^-0.5 * log2(e)

  // Q fragments for the whole row tile (held in registers)
  v16bf qf[4];
#pragma unroll
  for (int c = 0; c < 4; ++c)
    qf[c] = load_bf16_frag(Qb + (size_t)(b * SEQ + q0 + n) * DK + c * 32 + kh * 8);

  v8f acc[8];
  float m_i[8], l_i[8];
#pragma unroll
  for (int t = 0; t < 8; ++t) acc[t] = zero8();
#pragma unroll
  for (int g = 0; g < 8; ++g) { m_i[g] = -__builtin_inff(); l_i[g] = 0.0f; }

  const __bf16* Kbase = Kb + (size_t)b * SEQ * DK;
  const __bf16* Vbase = Vt + (size_t)b * DK * SEQ;

  for (int j0 = 0; j0 < SEQ; j0 += 32) {
    // S = Q K^T for two 16-key tiles (8 WMMAs)
    v8f st0 = zero8(), st1 = zero8();
#pragma unroll
    for (int c = 0; c < 4; ++c) {
      v16bf kf0 = load_bf16_frag(Kbase + (size_t)(j0 + n) * DK + c * 32 + kh * 8);
      st0 = __builtin_amdgcn_wmma_f32_16x16x32_bf16(false, qf[c], false, kf0,
                                                    (short)0, st0, false, false);
      v16bf kf1 = load_bf16_frag(Kbase + (size_t)(j0 + 16 + n) * DK + c * 32 + kh * 8);
      st1 = __builtin_amdgcn_wmma_f32_16x16x32_bf16(false, qf[c], false, kf1,
                                                    (short)0, st1, false, false);
    }

    // online softmax (base-2 domain); C-layout: VGPR g <-> row g (+8 for upper half)
    float alpha[8];
#pragma unroll
    for (int g = 0; g < 8; ++g) {
      float r = fmaxf(st0[g], st1[g]) * sc;
      r = fmaxf(r, __shfl_xor(r, 1, 32));
      r = fmaxf(r, __shfl_xor(r, 2, 32));
      r = fmaxf(r, __shfl_xor(r, 4, 32));
      r = fmaxf(r, __shfl_xor(r, 8, 32));
      float mn = fmaxf(m_i[g], r);
      alpha[g] = __builtin_exp2f(m_i[g] - mn);
      m_i[g]   = mn;

      float p0 = __builtin_exp2f(st0[g] * sc - mn);
      float p1 = __builtin_exp2f(st1[g] * sc - mn);
      int row = g + (kh ? 8 : 0);
      lds_p[wave][row][n]      = (__bf16)p0;   // ds_store_b16
      lds_p[wave][row][16 + n] = (__bf16)p1;

      float rs = p0 + p1;
      rs += __shfl_xor(rs, 1, 32);
      rs += __shfl_xor(rs, 2, 32);
      rs += __shfl_xor(rs, 4, 32);
      rs += __shfl_xor(rs, 8, 32);
      l_i[g] = l_i[g] * alpha[g] + rs;
    }

#pragma unroll
    for (int t = 0; t < 8; ++t)
#pragma unroll
      for (int g = 0; g < 8; ++g) acc[t][g] *= alpha[g];

    // P: C-layout -> A-layout via LDS (wave-private, in-order DS + explicit wait)
    asm volatile("s_wait_dscnt 0" ::: "memory");
    v16bf pf = load_bf16_frag(&lds_p[wave][n][kh * 8]);

    // O += P V for 8 column tiles (8 WMMAs); V^T rows give contiguous B-frags
#pragma unroll
    for (int t = 0; t < 8; ++t) {
      v16bf vf = load_bf16_frag(Vbase + (size_t)(t * 16 + n) * SEQ + j0 + kh * 8);
      acc[t] = __builtin_amdgcn_wmma_f32_16x16x32_bf16(false, pf, false, vf,
                                                       (short)0, acc[t], false, false);
    }
  }

  // normalize and write fp32 output [NB][SEQ][DK]
  float inv[8];
#pragma unroll
  for (int g = 0; g < 8; ++g) inv[g] = 1.0f / l_i[g];
  float* Orow = Out + (size_t)(b * SEQ + q0) * DK;
#pragma unroll
  for (int t = 0; t < 8; ++t)
#pragma unroll
    for (int g = 0; g < 8; ++g)
      Orow[(size_t)(g + (kh ? 8 : 0)) * DK + t * 16 + n] = acc[t][g] * inv[g];
}

// ---------- launcher ----------
extern "C" void kernel_launch(void* const* d_in, const int* in_sizes, int n_in,
                              void* d_out, int out_size, void* d_ws, size_t ws_size,
                              hipStream_t stream) {
  const float* X  = (const float*)d_in[0];
  const float* Wq = (const float*)d_in[1];
  const float* bq = (const float*)d_in[2];
  const float* Wk = (const float*)d_in[3];
  const float* bk = (const float*)d_in[4];
  const float* Wv = (const float*)d_in[5];
  const float* bv = (const float*)d_in[6];
  float* Out = (float*)d_out;

  // workspace layout (bf16): Wt[3][DK][DIM] | Qb[ROWS][DK] | Kb[ROWS][DK] | Vt[NB][DK][SEQ]
  __bf16* ws = (__bf16*)d_ws;
  __bf16* Wt = ws;
  __bf16* Qb = ws + (size_t)3 * DK * DIM;
  __bf16* Kb = Qb + (size_t)ROWS * DK;
  __bf16* Vt = Kb + (size_t)ROWS * DK;
  (void)in_sizes; (void)n_in; (void)out_size; (void)ws_size;

  dim3 cvg((DK * DIM) / 256);
  wcvt_kernel<<<cvg, 256, 0, stream>>>(Wq, Wt);
  wcvt_kernel<<<cvg, 256, 0, stream>>>(Wk, Wt + (size_t)DK * DIM);
  wcvt_kernel<<<cvg, 256, 0, stream>>>(Wv, Wt + (size_t)2 * DK * DIM);

  qkv_proj_kernel<<<dim3(ROWS / 128, 3), 256, 0, stream>>>(X, Wt, bq, bk, bv,
                                                           Qb, Kb, Vt);

  flash_attn_kernel<<<dim3(SEQ / (FA_WAVES * 16), NB), FA_WAVES * 32, 0, stream>>>(
      Qb, Kb, Vt, Out);
}
